// MultiheadSelfAttention_12214886990248
// MI455X (gfx1250) — matrix operand order repaired and checked by reading, not compiled, not run on hardware
//
#include <hip/hip_runtime.h>
#include <hip/hip_bf16.h>
#include <math.h>
#include <stdint.h>

#define S_LEN 4096
#define EMB   1024
#define NH    16
#define HD    64

typedef __attribute__((ext_vector_type(16))) __bf16 v16bf;
typedef __attribute__((ext_vector_type(8)))  __bf16 v8bf;
typedef __attribute__((ext_vector_type(4)))  __bf16 v4bf;
typedef __attribute__((ext_vector_type(8)))  float  v8f;

union Frag { v16bf v; v8bf h[2]; };

__device__ __forceinline__ v8f wmma_bf16(const Frag& a, const Frag& b, v8f c) {
    // D = A(16x32 bf16) x B(32x16 bf16) + C(16x16 f32)
    return __builtin_amdgcn_wmma_f32_16x16x32_bf16(
        /*neg_a=*/false, a.v, /*neg_b=*/false, b.v,
        /*c_mod=*/(short)0, c, /*reuse_a=*/false, /*reuse_b=*/false);
}

// ---- CDNA5 async global->LDS staging (ASYNCcnt path), with sync fallback ----
#if defined(__AMDGCN__) && __has_builtin(__builtin_amdgcn_global_load_async_to_lds_b128)
#define USE_ASYNC 1
#else
#define USE_ASYNC 0
#if defined(__AMDGCN__)
#warning "gfx1250 global_load_async_to_lds_b128 builtin unavailable; using sync staging"
#endif
#endif

#if USE_ASYNC
typedef __attribute__((ext_vector_type(4))) int v4i;
typedef __attribute__((address_space(1))) v4i GV4;   // int4 in global AS
typedef __attribute__((address_space(3))) v4i LV4;   // int4 in LDS AS
// generic->AS casts routed through integers (low 32 bits of generic LDS ptr = LDS offset)
#define ASYNC_CP16(gsrc, ldst)                                   \
    __builtin_amdgcn_global_load_async_to_lds_b128(              \
        (GV4*)(uintptr_t)(gsrc),                                 \
        (LV4*)(uint32_t)(uintptr_t)(ldst), 0, 0)
#if __has_builtin(__builtin_amdgcn_s_wait_asynccnt)
#define WAIT_ASYNC(n) __builtin_amdgcn_s_wait_asynccnt(n)
#else
#define WAIT_ASYNC(n) asm volatile("s_wait_asynccnt %0" :: "i"(n) : "memory")
#endif
#define STAGE_CP16(gsrc, ldst) ASYNC_CP16(gsrc, ldst)
#else
#define WAIT_ASYNC(n)
#define STAGE_CP16(gsrc, ldst) (*(v8bf*)(ldst) = *(const v8bf*)(gsrc))
#endif

// ---------------- conversion kernels ----------------

__global__ void cvt_f32_bf16(const float* __restrict__ in, __bf16* __restrict__ out, int n) {
    int i = (blockIdx.x * blockDim.x + threadIdx.x) * 4;
    if (i < n) {
        float4 f = *(const float4*)(in + i);
        v4bf o;
        o[0] = (__bf16)f.x; o[1] = (__bf16)f.y; o[2] = (__bf16)f.z; o[3] = (__bf16)f.w;
        *(v4bf*)(out + i) = o;
    }
}

// wt[n][k] = (bf16) w[k][n]   (transpose so WMMA B-columns are contiguous)
__global__ void cvt_w_t(const float* __restrict__ w, __bf16* __restrict__ wt) {
    int n = blockIdx.x * blockDim.x + threadIdx.x;   // fast dim, coalesced read
    int k = blockIdx.y;
    wt[(size_t)n * EMB + k] = (__bf16)w[(size_t)k * EMB + n];
}

// ------- projection GEMM: C[S,E] = A[S,E] x Bt[E,E]^T (Bt pre-transposed) -------
// 128-thread block = 4 waves (2x2), 64x64 tile per wave, 128x128 per block.
// Double-buffered LDS, async global->LDS staging overlapped with WMMA.

#define BM 128
#define BN 128
#define BK 32
#define GSTR (BK + 8)   // LDS row stride 40 bf16 = 80B

__global__ __launch_bounds__(128) void gemm_proj(const __bf16* __restrict__ A,
                                                 const __bf16* __restrict__ Bt,
                                                 __bf16* __restrict__ C,
                                                 float scale, int transposed) {
    __shared__ __bf16 sA[2][BM][GSTR];
    __shared__ __bf16 sB[2][BN][GSTR];

    const int tid  = threadIdx.x;
    const int lane = tid & 31, wid = tid >> 5;
    const int lh   = lane & 15, hi = lane >> 4;
    const int wm   = wid >> 1,  wn = wid & 1;     // 2x2 wave grid, 64x64 per wave
    const int m0   = blockIdx.y * BM, n0 = blockIdx.x * BN;

    v8f acc[4][4];
#pragma unroll
    for (int mt = 0; mt < 4; ++mt)
#pragma unroll
        for (int nt = 0; nt < 4; ++nt)
#pragma unroll
            for (int i = 0; i < 8; ++i) acc[mt][nt][i] = 0.f;

    // stage k-slice k0 into buffer buf: 512 chunks of 16B per array, 4/thread
    auto stage = [&](int buf, int k0) {
#pragma unroll
        for (int c = tid; c < BM * (BK / 8); c += 128) {
            int row = c >> 2, kc = (c & 3) * 8;
            STAGE_CP16(&A [(size_t)(m0 + row) * EMB + k0 + kc], &sA[buf][row][kc]);
            STAGE_CP16(&Bt[(size_t)(n0 + row) * EMB + k0 + kc], &sB[buf][row][kc]);
        }
    };

    const int NIT = EMB / BK;
    stage(0, 0);                                    // prologue (8 async ops/wave)
    for (int it = 0; it < NIT; ++it) {
        const int buf = it & 1;
        __syncthreads();                            // reads of buf^1 finished
        if (it + 1 < NIT) {
            stage(buf ^ 1, (it + 1) * BK);
            WAIT_ASYNC(8);                          // current buffer's 8 ops done
        } else {
            WAIT_ASYNC(0);
        }
        __syncthreads();                            // all waves' stage complete

        Frag af[4], bfr[4];
#pragma unroll
        for (int mt = 0; mt < 4; ++mt) {
            int row = wm * 64 + mt * 16 + lh, kb = hi * 8;
            af[mt].h[0] = *(const v8bf*)&sA[buf][row][kb];        // K = kb..kb+7
            af[mt].h[1] = *(const v8bf*)&sA[buf][row][16 + kb];   // K = 16+kb..
        }
#pragma unroll
        for (int nt = 0; nt < 4; ++nt) {
            int col = wn * 64 + nt * 16 + lh, kb2 = hi * 8;
            bfr[nt].h[0] = *(const v8bf*)&sB[buf][col][kb2];      // K = kb2..+7
            bfr[nt].h[1] = *(const v8bf*)&sB[buf][col][16 + kb2]; // K = 16+kb2..
        }
#pragma unroll
        for (int mt = 0; mt < 4; ++mt)
#pragma unroll
            for (int nt = 0; nt < 4; ++nt)
                acc[mt][nt] = wmma_bf16(af[mt], bfr[nt], acc[mt][nt]);
    }

#pragma unroll
    for (int mt = 0; mt < 4; ++mt)
#pragma unroll
        for (int nt = 0; nt < 4; ++nt)
#pragma unroll
            for (int i = 0; i < 8; ++i) {
                float v = acc[mt][nt][i] * scale;
                int r  = m0 + wm * 64 + mt * 16 + hi * 8 + i;   // C row = vgpr + 8*hi
                int cc = n0 + wn * 64 + nt * 16 + lh;           // C col = lane&15
                if (transposed) C[(size_t)cc * S_LEN + r] = (__bf16)v;  // Vt[E][S]
                else            C[(size_t)r * EMB + cc]   = (__bf16)v;
            }
}

// ---------------- flash attention: per (head, 128-query block) ----------------
// 8 waves x 16 query rows; 64-key tiles, double-buffered async K/V staging.

#define QB 128     // queries per block
#define KTB 64     // keys per iteration
#define APAD 8     // LDS row stride = 72 bf16 = 144B

__global__ __launch_bounds__(256) void attn(const __bf16* __restrict__ Q,
                                            const __bf16* __restrict__ K,
                                            const __bf16* __restrict__ Vt,
                                            float* __restrict__ out) {
    __shared__ __bf16 sK[2][KTB][HD + APAD];     // keys x head-dim
    __shared__ __bf16 sV[2][HD][KTB + APAD];     // head-dim x keys (from Vt)
    __shared__ __bf16 sP[8][16][KTB + APAD];     // per-wave probabilities

    const int tid  = threadIdx.x;
    const int lane = tid & 31, wid = tid >> 5;
    const int lh   = lane & 15, hi = lane >> 4;
    const int head = blockIdx.y;
    const int s0   = blockIdx.x * QB;
    const int srow = s0 + wid * 16 + lh;

    // Q fragments for this wave's 16 rows (d = 0..63 -> two 16x32 A-frags)
    Frag qf[2];
    {
        const __bf16* qp = Q + (size_t)srow * EMB + head * HD;
        int kb = hi * 8;
        qf[0].h[0] = *(const v8bf*)(qp + kb);
        qf[0].h[1] = *(const v8bf*)(qp + 16 + kb);
        qf[1].h[0] = *(const v8bf*)(qp + 32 + kb);
        qf[1].h[1] = *(const v8bf*)(qp + 48 + kb);
    }

    v8f accO[4];
    float m_i[8], l_i[8];
#pragma unroll
    for (int dt = 0; dt < 4; ++dt)
#pragma unroll
        for (int i = 0; i < 8; ++i) accO[dt][i] = 0.f;
#pragma unroll
    for (int i = 0; i < 8; ++i) { m_i[i] = -1e30f; l_i[i] = 0.f; }

    // stage key-tile t0 into buffer buf: 512 16B chunks over K and V, 2/thread
    auto stage = [&](int buf, int t0) {
#pragma unroll
        for (int c = tid; c < KTB * (HD / 8); c += 256) {
            int r = c >> 3, cc = (c & 7) * 8;
            STAGE_CP16(&K [(size_t)(t0 + r) * EMB + head * HD + cc],  &sK[buf][r][cc]);
            STAGE_CP16(&Vt[(size_t)(head * HD + r) * S_LEN + t0 + cc], &sV[buf][r][cc]);
        }
    };

    const int NIT = S_LEN / KTB;
    stage(0, 0);                                    // prologue (4 async ops/wave)
    for (int it = 0; it < NIT; ++it) {
        const int buf = it & 1;
        __syncthreads();                            // prev reads of buf^1 done
        if (it + 1 < NIT) {
            stage(buf ^ 1, (it + 1) * KTB);
            WAIT_ASYNC(4);                          // current buffer staged
        } else {
            WAIT_ASYNC(0);
        }
        __syncthreads();

        // scores S = Q x K^T (already scaled by 0.125*log2e via Q)
        v8f sc[4];
#pragma unroll
        for (int nt = 0; nt < 4; ++nt) {
            int t = nt * 16 + lh, kb2 = hi * 8;
            Frag b0, b1;
            b0.h[0] = *(const v8bf*)&sK[buf][t][kb2];
            b0.h[1] = *(const v8bf*)&sK[buf][t][16 + kb2];
            b1.h[0] = *(const v8bf*)&sK[buf][t][32 + kb2];
            b1.h[1] = *(const v8bf*)&sK[buf][t][48 + kb2];
            v8f z;
#pragma unroll
            for (int i = 0; i < 8; ++i) z[i] = 0.f;
            z = wmma_bf16(qf[0], b0, z);
            sc[nt] = wmma_bf16(qf[1], b1, z);
        }

        // online softmax (base-2); rows live per-vgpr, cols across 16-lane halves
        float mnew[8], corr[8], rs[8];
#pragma unroll
        for (int i = 0; i < 8; ++i) {
            float v = fmaxf(fmaxf(sc[0][i], sc[1][i]), fmaxf(sc[2][i], sc[3][i]));
            v = fmaxf(v, __shfl_xor(v, 1, 32));
            v = fmaxf(v, __shfl_xor(v, 2, 32));
            v = fmaxf(v, __shfl_xor(v, 4, 32));
            v = fmaxf(v, __shfl_xor(v, 8, 32));
            mnew[i] = fmaxf(m_i[i], v);
            corr[i] = exp2f(m_i[i] - mnew[i]);
            m_i[i]  = mnew[i];
            rs[i]   = 0.f;
        }
#pragma unroll
        for (int nt = 0; nt < 4; ++nt)
#pragma unroll
            for (int i = 0; i < 8; ++i) {
                float p = exp2f(sc[nt][i] - mnew[i]);
                rs[i] += p;
                sP[wid][hi * 8 + i][nt * 16 + lh] = (__bf16)p;
            }
#pragma unroll
        for (int i = 0; i < 8; ++i) {
            float r = rs[i];
            r += __shfl_xor(r, 1, 32);
            r += __shfl_xor(r, 2, 32);
            r += __shfl_xor(r, 4, 32);
            r += __shfl_xor(r, 8, 32);
            l_i[i] = l_i[i] * corr[i] + r;
#pragma unroll
            for (int dt = 0; dt < 4; ++dt) accO[dt][i] *= corr[i];
        }
        __syncthreads();   // sP visible

        // O += P x V
#pragma unroll
        for (int tk = 0; tk < 2; ++tk) {
            Frag afr;
            int kb = hi * 8;
            afr.h[0] = *(const v8bf*)&sP[wid][lh][tk * 32 + kb];
            afr.h[1] = *(const v8bf*)&sP[wid][lh][tk * 32 + 16 + kb];
#pragma unroll
            for (int dt = 0; dt < 4; ++dt) {
                int d = dt * 16 + lh, kb2 = hi * 8;
                Frag bv;
                bv.h[0] = *(const v8bf*)&sV[buf][d][tk * 32 + kb2];
                bv.h[1] = *(const v8bf*)&sV[buf][d][tk * 32 + 16 + kb2];
                accO[dt] = wmma_bf16(afr, bv, accO[dt]);
            }
        }
    }

    // normalize and store f32 output [S, E]
#pragma unroll
    for (int dt = 0; dt < 4; ++dt)
#pragma unroll
        for (int i = 0; i < 8; ++i) {
            int r  = s0 + wid * 16 + hi * 8 + i;
            int cc = head * HD + dt * 16 + lh;
            out[(size_t)r * EMB + cc] = accO[dt][i] / l_i[i];
        }
}

// ---------------- launcher ----------------

extern "C" void kernel_launch(void* const* d_in, const int* in_sizes, int n_in,
                              void* d_out, int out_size, void* d_ws, size_t ws_size,
                              hipStream_t stream) {
    const float* x  = (const float*)d_in[0];
    const float* Wq = (const float*)d_in[1];
    const float* Wk = (const float*)d_in[2];
    const float* Wv = (const float*)d_in[3];
    float* out = (float*)d_out;

    char* ws = (char*)d_ws;
    __bf16* xb   = (__bf16*)(ws);                        //  8 MiB
    __bf16* wqt  = (__bf16*)(ws + ((size_t) 8 << 20));   //  2 MiB
    __bf16* wkt  = (__bf16*)(ws + ((size_t)10 << 20));   //  2 MiB
    __bf16* wvt  = (__bf16*)(ws + ((size_t)12 << 20));   //  2 MiB
    __bf16* qbuf = (__bf16*)(ws + ((size_t)14 << 20));   //  8 MiB
    __bf16* kbuf = (__bf16*)(ws + ((size_t)22 << 20));   //  8 MiB
    __bf16* vtb  = (__bf16*)(ws + ((size_t)30 << 20));   //  8 MiB (Vt[E][S])

    cvt_f32_bf16<<<(S_LEN * EMB) / (256 * 4), 256, 0, stream>>>(x, xb, S_LEN * EMB);
    dim3 gw(EMB / 256, EMB);
    cvt_w_t<<<gw, 256, 0, stream>>>(Wq, wqt);
    cvt_w_t<<<gw, 256, 0, stream>>>(Wk, wkt);
    cvt_w_t<<<gw, 256, 0, stream>>>(Wv, wvt);

    const float SCALE_Q = 0.125f * 1.4426950408889634f;  // 1/sqrt(64) * log2(e)
    dim3 gg(EMB / BN, S_LEN / BM);
    gemm_proj<<<gg, 128, 0, stream>>>(xb, wqt, qbuf, SCALE_Q, 0);
    gemm_proj<<<gg, 128, 0, stream>>>(xb, wkt, kbuf, 1.0f, 0);
    gemm_proj<<<gg, 128, 0, stream>>>(xb, wvt, vtb, 1.0f, 1);   // Vt[E][S]

    dim3 ga(S_LEN / QB, NH);
    attn<<<ga, 256, 0, stream>>>(qbuf, kbuf, vtb, out);

    (void)in_sizes; (void)n_in; (void)out_size; (void)ws_size;
}
